// PraxisBlock_24567212933863
// MI455X (gfx1250) — matrix operand order repaired and checked by reading, not compiled, not run on hardware
//
#include <hip/hip_runtime.h>
#include <hip/hip_bf16.h>

#define D_MODEL 1024
#define N_HEAD  16
#define D_HEAD  64
#define N_EXP   8
#define TOPK    2
#define FF_DIM  4096
#define SEQ     1024
#define BN      512   // block N-tile (4 WMMA tiles per wave)

typedef __attribute__((ext_vector_type(16))) __bf16 v16bf;
typedef __attribute__((ext_vector_type(8)))  float  v8f;

union ABf { v16bf v; unsigned short h[16]; uint4 q[2]; };

__device__ inline unsigned short f2bf(float f) {
  union { float f; unsigned u; } x; x.f = f;
  unsigned r = x.u + 0x7FFFu + ((x.u >> 16) & 1u);
  return (unsigned short)(r >> 16);
}
__device__ inline float bf2f(unsigned short h) {
  union { unsigned u; float f; } x; x.u = ((unsigned)h) << 16; return x.f;
}
__device__ inline float gelu_tanh(float x) {
  float x3 = x * x * x;
  return 0.5f * x * (1.0f + tanhf(0.7978845608028654f * (x + 0.044715f * x3)));
}

__device__ inline v8f wmma_bf16(const ABf& a, const ABf& b, v8f c) {
  return __builtin_amdgcn_wmma_f32_16x16x32_bf16(false, a.v, false, b.v,
                                                 (short)0, c, false, false);
}

// ---- CDNA5 async global->LDS staging (ISA 08_async_tensor: VGLOBAL op 98) ----
__device__ inline void async_b128(unsigned ldsoff, unsigned long long ga) {
  asm volatile("global_load_async_to_lds_b128 %0, %1, off"
               :: "v"(ldsoff), "v"(ga) : "memory");
}
__device__ inline void wait_async0() {
  asm volatile("s_wait_asynccnt 0x0" ::: "memory");
}

// ---- fragment loaders (contiguous k = 16*(l/16)+s order, same for A and B, wave32) ----
// A 16x32: lane m=l&15, kb=(l>=16)?16:0; elements = k[kb..kb+15]
__device__ inline void load_a_lane(ABf& a, const unsigned short* rp) {
  int kb = ((threadIdx.x & 31) >= 16) ? 16 : 0;
  a.q[0] = *(const uint4*)(rp + kb);
  a.q[1] = *(const uint4*)(rp + kb + 8);
}
__device__ inline void load_a(ABf& a, const unsigned short* base, int ld, int col0) {
  int m = threadIdx.x & 15;
  load_a_lane(a, base + (size_t)m * ld + col0);
}
// B 32x16: lane n=l&15, kb=(l>=16)?16:0; 16 contiguous bf16 with NxK storage.
__device__ inline void load_frag16(ABf& b, const unsigned short* p) {
  b.q[0] = *(const uint4*)(p);
  b.q[1] = *(const uint4*)(p + 8);
}
__device__ inline const unsigned short* frag_ptr(const unsigned short* NT, size_t ldk,
                                                 int n0) {
  int lane = threadIdx.x & 31;
  return NT + (size_t)(n0 + (lane & 15)) * ldk + ((lane >= 16) ? 16 : 0);
}

// stage BN (n-rows) x 32 (k) bf16 chunk of an NxK matrix into LDS (32KB),
// async global->LDS b128; caller then wait_async0 + barrier. 2 rows/thread.
__device__ inline void stage_bt(const unsigned short* BTb, int Kd, int k,
                                unsigned ldsbase) {
  int r0 = threadIdx.x * 2;
#pragma unroll
  for (int j = 0; j < 2; ++j) {
    int row = r0 + j;
    const unsigned short* g = BTb + (size_t)row * Kd + k;
    unsigned l = ldsbase + (unsigned)(row * 64);
#pragma unroll
    for (int s = 0; s < 4; ++s)
      async_b128(l + 16u * s, (unsigned long long)(size_t)(g + 8 * s));
  }
}

// ======================= kernels =======================

// dst (N x K, bf16) = transpose(src (K x N, f32)); one matrix per blockIdx.y;
// K = 1<<kshift (power of two) so index decode is shift/mask.
__global__ void convert_transpose_k(const float* __restrict__ src,
                                    unsigned short* __restrict__ dst,
                                    int kshift, int N) {
  int K = 1 << kshift;
  size_t mn = (size_t)K * N;
  const float* s = src + (size_t)blockIdx.y * mn;
  unsigned short* d = dst + (size_t)blockIdx.y * mn;
  size_t stride = (size_t)gridDim.x * blockDim.x;
  for (size_t i = (size_t)blockIdx.x * blockDim.x + threadIdx.x; i < mn; i += stride) {
    int n = (int)(i >> kshift);
    int k = (int)(i & (size_t)(K - 1));
    d[i] = f2bf(s[(size_t)k * N + n]);
  }
}

__global__ void init_misc_k(int* fill, float* cntf, float* psum,
                            unsigned short* zrow) {
  int i = blockIdx.x * blockDim.x + threadIdx.x;
  if (i < N_EXP) { fill[i] = 0; cntf[i] = 0.f; psum[i] = 0.f; }
  for (int j = i; j < FF_DIM; j += gridDim.x * blockDim.x) zrow[j] = 0;
}

__global__ __launch_bounds__(256) void rmsnorm_bf16_k(
    const float* __restrict__ x, const float* __restrict__ w,
    unsigned short* __restrict__ out) {
  int row = blockIdx.x;
  const float* xr = x + (size_t)row * D_MODEL;
  __shared__ float red[256];
  float s = 0.f;
  for (int d = threadIdx.x; d < D_MODEL; d += 256) { float v = xr[d]; s += v * v; }
  red[threadIdx.x] = s; __syncthreads();
  for (int off = 128; off; off >>= 1) {
    if ((int)threadIdx.x < off) red[threadIdx.x] += red[threadIdx.x + off];
    __syncthreads();
  }
  float inv = rsqrtf(red[0] / (float)D_MODEL + 1e-6f);
  for (int d = threadIdx.x; d < D_MODEL; d += 256)
    out[(size_t)row * D_MODEL + d] = f2bf(xr[d] * inv * w[d]);
}

// block-cooperative GEMM, transposed-B (NxK bf16): block = 16 rows x 512 cols;
// each of 8 waves owns 4 accumulator tiles -> 4 WMMA per staged chunk.
__global__ __launch_bounds__(256) void gemm_coop_k(
    const unsigned short* __restrict__ A, const unsigned short* __restrict__ BT,
    int Mt, int Nt, int Kd, float scale,
    float* __restrict__ Cf, unsigned short* __restrict__ Cbf,
    unsigned short* __restrict__ CbfT, const float* __restrict__ addsrc) {
  __shared__ unsigned short btile[BN * 32];
  int nb = Nt / BN;
  int tm = blockIdx.x / nb;
  int tb = blockIdx.x - tm * nb;
  int wid = threadIdx.x >> 5;
  int lane = threadIdx.x & 31;

  const unsigned short* arow = A + (size_t)(tm * 16 + (lane & 15)) * Kd;
  const unsigned short* bt = BT + (size_t)tb * BN * Kd;
  unsigned ldsbase = (unsigned)(size_t)&btile[0];
  const unsigned short* fps[4];
#pragma unroll
  for (int j = 0; j < 4; ++j) fps[j] = frag_ptr(&btile[0], 32, wid * 64 + j * 16);

  v8f cc[4] = {};
  for (int k = 0; k < Kd; k += 32) {
    stage_bt(bt, Kd, k, ldsbase);
    __builtin_prefetch(arow + k + 32, 0, 1);
    wait_async0();
    __syncthreads();
    ABf a, b;
    load_a_lane(a, arow + k);
#pragma unroll
    for (int j = 0; j < 4; ++j) {
      load_frag16(b, fps[j]);
      cc[j] = wmma_bf16(a, b, cc[j]);
    }
    __syncthreads();
  }
  int mb = tm * 16 + ((lane >= 16) ? 8 : 0);
  int nbase = tb * BN + wid * 64 + (lane & 15);
#pragma unroll
  for (int j = 0; j < 4; ++j) {
    int n = nbase + j * 16;
#pragma unroll
    for (int r = 0; r < 8; ++r) {
      size_t idx = (size_t)(mb + r) * Nt + n;
      float v = cc[j][r] * scale;
      if (addsrc) v += addsrc[idx];
      if (Cf)   Cf[idx] = v;
      if (Cbf)  Cbf[idx] = f2bf(v);
      if (CbfT) CbfT[(size_t)n * Mt + (mb + r)] = f2bf(v);
    }
  }
}

// flash attention: one wave per (b, head, 16-query tile); KV blocks of 32.
// K row-major bf16 [token][D]; V transposed bf16 [Dcol][token].
__global__ __launch_bounds__(256) void attn_k(
    const unsigned short* __restrict__ Q, const unsigned short* __restrict__ Km,
    const unsigned short* __restrict__ Vt, unsigned short* __restrict__ Om,
    int Mt, int nWaves) {
  __shared__ unsigned short pbuf[8][16 * 32];
  int wid = threadIdx.x >> 5;
  int gw = blockIdx.x * 8 + wid;
  if (gw >= nWaves) return;
  const int QT = SEQ / 16;
  int qt = gw % QT;
  int bh = gw / QT;
  int h = bh % N_HEAD, b = bh / N_HEAD;
  int lane = threadIdx.x & 31;
  bool hi = lane >= 16;
  int nn = lane & 15;
  int kfo = hi ? 16 : 0;

  const unsigned short* qbase = Q + ((size_t)(b * SEQ + qt * 16)) * D_MODEL + h * D_HEAD;
  ABf qa0, qa1;
  load_a(qa0, qbase, D_MODEL, 0);
  load_a(qa1, qbase, D_MODEL, 32);

  v8f o0 = {}, o1 = {}, o2 = {}, o3 = {};
  float mrow[8], lrow[8];
#pragma unroll
  for (int r = 0; r < 8; ++r) { mrow[r] = -1e30f; lrow[r] = 0.f; }

  int nblocks = (qt * 16 + 15) / 32 + 1;
  for (int kb = 0; kb < nblocks; ++kb) {
    int key0 = kb * 32;
    v8f s0 = {}, s1 = {};
    {
      ABf bk;
      const unsigned short* k0p =
          Km + ((size_t)(b * SEQ + key0 + nn)) * D_MODEL + h * D_HEAD;
      const unsigned short* k1p =
          Km + ((size_t)(b * SEQ + key0 + 16 + nn)) * D_MODEL + h * D_HEAD;
      load_frag16(bk, k0p + kfo);       s0 = wmma_bf16(qa0, bk, s0);
      load_frag16(bk, k0p + 32 + kfo);  s0 = wmma_bf16(qa1, bk, s0);
      load_frag16(bk, k1p + kfo);       s1 = wmma_bf16(qa0, bk, s1);
      load_frag16(bk, k1p + 32 + kfo);  s1 = wmma_bf16(qa1, bk, s1);
    }
#pragma unroll
    for (int r = 0; r < 8; ++r) {
      int mg = qt * 16 + r + (hi ? 8 : 0);
      float sc0 = (key0 + nn      <= mg) ? s0[r] : -1e30f;
      float sc1 = (key0 + 16 + nn <= mg) ? s1[r] : -1e30f;
      float v = fmaxf(sc0, sc1);
      for (int off = 1; off < 16; off <<= 1) v = fmaxf(v, __shfl_xor(v, off, 32));
      float nm = fmaxf(mrow[r], v);
      float alpha = __expf(mrow[r] - nm);
      mrow[r] = nm;
      float p0 = __expf(sc0 - nm);
      float p1 = __expf(sc1 - nm);
      float rs = p0 + p1;
      for (int off = 1; off < 16; off <<= 1) rs += __shfl_xor(rs, off, 32);
      lrow[r] = lrow[r] * alpha + rs;
      o0[r] *= alpha; o1[r] *= alpha; o2[r] *= alpha; o3[r] *= alpha;
      int mloc = r + (hi ? 8 : 0);
      pbuf[wid][mloc * 32 + nn]      = f2bf(p0);
      pbuf[wid][mloc * 32 + 16 + nn] = f2bf(p1);
    }
    ABf pa;
    load_a(pa, &pbuf[wid][0], 32, 0);  // same-wave LDS RAW; compiler inserts waits
    const unsigned short* vtb =
        Vt + (size_t)(h * D_HEAD) * Mt + (size_t)(b * SEQ + key0 + kfo);
    ABf vb;
    load_frag16(vb, vtb + (size_t)(nn)      * Mt); o0 = wmma_bf16(pa, vb, o0);
    load_frag16(vb, vtb + (size_t)(nn + 16) * Mt); o1 = wmma_bf16(pa, vb, o1);
    load_frag16(vb, vtb + (size_t)(nn + 32) * Mt); o2 = wmma_bf16(pa, vb, o2);
    load_frag16(vb, vtb + (size_t)(nn + 48) * Mt); o3 = wmma_bf16(pa, vb, o3);
  }
  size_t obase = ((size_t)(b * SEQ + qt * 16)) * D_MODEL + h * D_HEAD;
#pragma unroll
  for (int r = 0; r < 8; ++r) {
    int mloc = r + (hi ? 8 : 0);
    float linv = 1.0f / lrow[r];
    size_t rowoff = obase + (size_t)mloc * D_MODEL;
    Om[rowoff + 0  + nn] = f2bf(o0[r] * linv);
    Om[rowoff + 16 + nn] = f2bf(o1[r] * linv);
    Om[rowoff + 32 + nn] = f2bf(o2[r] * linv);
    Om[rowoff + 48 + nn] = f2bf(o3[r] * linv);
  }
}

// one wave per token: logits, softmax, top-2, expert lists
__global__ __launch_bounds__(256) void router_k(
    const unsigned short* __restrict__ Hb, const float* __restrict__ Rw,
    int* fill, int* lists, int* slots, float* gatesb,
    float* cntf, float* psum, int Mtok) {
  int t = (int)((blockIdx.x * blockDim.x + threadIdx.x) >> 5);
  if (t >= Mtok) return;
  int lane = threadIdx.x & 31;
  float acc[N_EXP];
#pragma unroll
  for (int e = 0; e < N_EXP; ++e) acc[e] = 0.f;
  const unsigned short* hrow = Hb + (size_t)t * D_MODEL;
  for (int d = lane; d < D_MODEL; d += 32) {
    float hv = bf2f(hrow[d]);
#pragma unroll
    for (int e = 0; e < N_EXP; ++e) acc[e] += hv * Rw[d * N_EXP + e];
  }
#pragma unroll
  for (int e = 0; e < N_EXP; ++e)
    for (int off = 16; off >= 1; off >>= 1) acc[e] += __shfl_xor(acc[e], off, 32);
  if (lane == 0) {
    float mx = acc[0];
#pragma unroll
    for (int e = 1; e < N_EXP; ++e) mx = fmaxf(mx, acc[e]);
    float p[N_EXP], s = 0.f;
#pragma unroll
    for (int e = 0; e < N_EXP; ++e) { p[e] = __expf(acc[e] - mx); s += p[e]; }
    float sinv = 1.0f / s;
#pragma unroll
    for (int e = 0; e < N_EXP; ++e) { p[e] *= sinv; atomicAdd(&psum[e], p[e]); }
    int i0 = 0;
#pragma unroll
    for (int e = 1; e < N_EXP; ++e) if (p[e] > p[i0]) i0 = e;
    int i1 = (i0 == 0) ? 1 : 0;
#pragma unroll
    for (int e = 0; e < N_EXP; ++e) if (e != i0 && p[e] > p[i1]) i1 = e;
    int s0 = atomicAdd(&fill[i0], 1);
    lists[i0 * Mtok + s0] = t; slots[i0 * Mtok + s0] = t * TOPK + 0;
    gatesb[i0 * Mtok + s0] = p[i0]; atomicAdd(&cntf[i0], 1.0f);
    int s1 = atomicAdd(&fill[i1], 1);
    lists[i1 * Mtok + s1] = t; slots[i1 * Mtok + s1] = t * TOPK + 1;
    gatesb[i1 * Mtok + s1] = p[i1]; atomicAdd(&cntf[i1], 1.0f);
  }
}

// MoE pass 1: act[slot] = gelu(gather(h) @ w1[e] + b1[e]); 4 tiles/wave
__global__ __launch_bounds__(256) void moe1_k(
    const unsigned short* __restrict__ Hb, const unsigned short* __restrict__ W1t,
    const float* __restrict__ B1, const int* __restrict__ lists,
    const int* __restrict__ slots, const int* __restrict__ fill,
    const unsigned short* __restrict__ zrow,
    unsigned short* __restrict__ act, int Mtok) {
  __shared__ unsigned short btile[BN * 32];
  int e = blockIdx.y;
  int Me = fill[e];
  const int nb = FF_DIM / BN;     // 8
  int tm = blockIdx.x / nb;
  int tb = blockIdx.x - tm * nb;
  if (tm * 16 >= Me) return;      // block-uniform exit
  int wid = threadIdx.x >> 5;
  int lane = threadIdx.x & 31;
  bool hi = lane >= 16;
  int rid = tm * 16 + (lane & 15);
  const unsigned short* arow =
      (rid < Me) ? (Hb + (size_t)lists[e * Mtok + rid] * D_MODEL) : zrow;

  const unsigned short* bt = W1t + (size_t)e * FF_DIM * D_MODEL +
                             (size_t)tb * BN * D_MODEL;
  unsigned ldsbase = (unsigned)(size_t)&btile[0];
  const unsigned short* fps[4];
#pragma unroll
  for (int j = 0; j < 4; ++j) fps[j] = frag_ptr(&btile[0], 32, wid * 64 + j * 16);

  v8f cc[4] = {};
  for (int k = 0; k < D_MODEL; k += 32) {
    stage_bt(bt, D_MODEL, k, ldsbase);
    wait_async0();
    __syncthreads();
    ABf a, b;
    load_a_lane(a, arow + k);
#pragma unroll
    for (int j = 0; j < 4; ++j) {
      load_frag16(b, fps[j]);
      cc[j] = wmma_bf16(a, b, cc[j]);
    }
    __syncthreads();
  }
  int nbase = tb * BN + wid * 64 + (lane & 15);
#pragma unroll
  for (int j = 0; j < 4; ++j) {
    int n = nbase + j * 16;
#pragma unroll
    for (int r = 0; r < 8; ++r) {
      int rw2 = tm * 16 + r + (hi ? 8 : 0);
      if (rw2 < Me) {
        float xv = cc[j][r] + B1[e * FF_DIM + n];
        int slot = slots[e * Mtok + rw2];
        act[(size_t)slot * FF_DIM + n] = f2bf(gelu_tanh(xv));
      }
    }
  }
}

// MoE pass 2: out[tok] += gate * (act[slot] @ w2[e] + b2[e]); 4 tiles/wave
__global__ __launch_bounds__(256) void moe2_k(
    const unsigned short* __restrict__ act, const unsigned short* __restrict__ W2t,
    const float* __restrict__ B2, const int* __restrict__ lists,
    const int* __restrict__ slots, const float* __restrict__ gatesb,
    const int* __restrict__ fill, const unsigned short* __restrict__ zrow,
    float* __restrict__ out, int Mtok) {
  __shared__ unsigned short btile[BN * 32];
  int e = blockIdx.y;
  int Me = fill[e];
  const int nb = D_MODEL / BN;    // 2
  int tm = blockIdx.x / nb;
  int tb = blockIdx.x - tm * nb;
  if (tm * 16 >= Me) return;      // block-uniform exit
  int wid = threadIdx.x >> 5;
  int lane = threadIdx.x & 31;
  bool hi = lane >= 16;
  int rid = tm * 16 + (lane & 15);
  const unsigned short* arow =
      (rid < Me) ? (act + (size_t)slots[e * Mtok + rid] * FF_DIM) : zrow;

  const unsigned short* bt = W2t + (size_t)e * D_MODEL * FF_DIM +
                             (size_t)tb * BN * FF_DIM;
  unsigned ldsbase = (unsigned)(size_t)&btile[0];
  const unsigned short* fps[4];
#pragma unroll
  for (int j = 0; j < 4; ++j) fps[j] = frag_ptr(&btile[0], 32, wid * 64 + j * 16);

  v8f cc[4] = {};
  for (int k = 0; k < FF_DIM; k += 32) {
    stage_bt(bt, FF_DIM, k, ldsbase);
    wait_async0();
    __syncthreads();
    ABf a, b;
    load_a_lane(a, arow + k);
#pragma unroll
    for (int j = 0; j < 4; ++j) {
      load_frag16(b, fps[j]);
      cc[j] = wmma_bf16(a, b, cc[j]);
    }
    __syncthreads();
  }
  int nbase = tb * BN + wid * 64 + (lane & 15);
#pragma unroll
  for (int j = 0; j < 4; ++j) {
    int n = nbase + j * 16;
#pragma unroll
    for (int r = 0; r < 8; ++r) {
      int rw2 = tm * 16 + r + (hi ? 8 : 0);
      if (rw2 < Me) {
        int tok = lists[e * Mtok + rw2];
        float g = gatesb[e * Mtok + rw2];
        float v = (cc[j][r] + B2[e * D_MODEL + n]) * g;
        atomicAdd(&out[(size_t)tok * D_MODEL + n], v);
      }
    }
  }
}

__global__ void loss_k(const float* cntf, const float* psum, float* tail, int Mtok) {
  if (threadIdx.x == 0) {
    float L = 0.f;
    for (int e = 0; e < N_EXP; ++e)
      L += (cntf[e] / (float)(Mtok * TOPK)) * (psum[e] / (float)Mtok);
    tail[0] = (float)N_EXP * L;
  }
  if ((int)threadIdx.x < N_EXP) tail[1 + threadIdx.x] = cntf[threadIdx.x];
}

// ======================= host launch =======================

extern "C" void kernel_launch(void* const* d_in, const int* in_sizes, int n_in,
                              void* d_out, int out_size, void* d_ws, size_t ws_size,
                              hipStream_t stream) {
  (void)n_in; (void)out_size; (void)ws_size;
  const float* x   = (const float*)d_in[0];
  const float* anw = (const float*)d_in[1];
  const float* wq  = (const float*)d_in[2];
  const float* wk  = (const float*)d_in[3];
  const float* wv  = (const float*)d_in[4];
  const float* wo  = (const float*)d_in[5];
  const float* mnw = (const float*)d_in[6];
  const float* rw  = (const float*)d_in[7];
  const float* w1  = (const float*)d_in[8];
  const float* b1  = (const float*)d_in[9];
  const float* w2  = (const float*)d_in[10];
  const float* b2  = (const float*)d_in[11];
  float* out = (float*)d_out;

  const int M = in_sizes[0] / D_MODEL;   // B*S tokens (2048)
  const int Bb = M / SEQ;

  // workspace layout
  char* wp = (char*)d_ws;
  unsigned short* xn   = (unsigned short*)wp; wp += (size_t)M * D_MODEL * 2;
  unsigned short* qb   = (unsigned short*)wp; wp += (size_t)M * D_MODEL * 2;
  unsigned short* kbuf = (unsigned short*)wp; wp += (size_t)M * D_MODEL * 2;
  unsigned short* vT   = (unsigned short*)wp; wp += (size_t)M * D_MODEL * 2;
  unsigned short* ao   = (unsigned short*)wp; wp += (size_t)M * D_MODEL * 2;
  unsigned short* hb   = (unsigned short*)wp; wp += (size_t)M * D_MODEL * 2;
  unsigned short* act  = (unsigned short*)wp; wp += (size_t)M * TOPK * FF_DIM * 2;
  unsigned short* wqt  = (unsigned short*)wp; wp += (size_t)D_MODEL * D_MODEL * 2;
  unsigned short* wkt  = (unsigned short*)wp; wp += (size_t)D_MODEL * D_MODEL * 2;
  unsigned short* wvt  = (unsigned short*)wp; wp += (size_t)D_MODEL * D_MODEL * 2;
  unsigned short* wot  = (unsigned short*)wp; wp += (size_t)D_MODEL * D_MODEL * 2;
  unsigned short* w1t  = (unsigned short*)wp; wp += (size_t)N_EXP * D_MODEL * FF_DIM * 2;
  unsigned short* w2t  = (unsigned short*)wp; wp += (size_t)N_EXP * FF_DIM * D_MODEL * 2;
  unsigned short* zrow = (unsigned short*)wp; wp += (size_t)FF_DIM * 2;
  int*   lists  = (int*)wp;   wp += (size_t)N_EXP * M * 4;
  int*   slots  = (int*)wp;   wp += (size_t)N_EXP * M * 4;
  float* gatesb = (float*)wp; wp += (size_t)N_EXP * M * 4;
  int*   fill   = (int*)wp;   wp += 64;
  float* cntf   = (float*)wp; wp += 64;
  float* psum   = (float*)wp; wp += 64;

  const float qscale = 0.125f; // 1/sqrt(64)

  init_misc_k<<<4, 256, 0, stream>>>(fill, cntf, psum, zrow);

  // one-time fp32 -> bf16 convert + transpose (weights become N x K)
  convert_transpose_k<<<dim3(512, 1), 256, 0, stream>>>(wq, wqt, 10, D_MODEL);
  convert_transpose_k<<<dim3(512, 1), 256, 0, stream>>>(wk, wkt, 10, D_MODEL);
  convert_transpose_k<<<dim3(512, 1), 256, 0, stream>>>(wv, wvt, 10, D_MODEL);
  convert_transpose_k<<<dim3(512, 1), 256, 0, stream>>>(wo, wot, 10, D_MODEL);
  convert_transpose_k<<<dim3(2048, N_EXP), 256, 0, stream>>>(w1, w1t, 10, FF_DIM);
  convert_transpose_k<<<dim3(2048, N_EXP), 256, 0, stream>>>(w2, w2t, 12, D_MODEL);

  // attn pre-norm
  rmsnorm_bf16_k<<<M, 256, 0, stream>>>(x, anw, xn);

  // Q (scaled), K, V projections; V stored transposed
  int pblocks = (M / 16) * (D_MODEL / BN);
  gemm_coop_k<<<pblocks, 256, 0, stream>>>(xn, wqt, M, D_MODEL, D_MODEL, qscale,
                                           (float*)0, qb, (unsigned short*)0,
                                           (const float*)0);
  gemm_coop_k<<<pblocks, 256, 0, stream>>>(xn, wkt, M, D_MODEL, D_MODEL, 1.0f,
                                           (float*)0, kbuf, (unsigned short*)0,
                                           (const float*)0);
  gemm_coop_k<<<pblocks, 256, 0, stream>>>(xn, wvt, M, D_MODEL, D_MODEL, 1.0f,
                                           (float*)0, (unsigned short*)0, vT,
                                           (const float*)0);

  // flash attention
  int nWaves = Bb * N_HEAD * (SEQ / 16);
  attn_k<<<(nWaves + 7) / 8, 256, 0, stream>>>(qb, kbuf, vT, ao, M, nWaves);

  // output projection + residual -> d_out holds x1
  gemm_coop_k<<<pblocks, 256, 0, stream>>>(ao, wot, M, D_MODEL, D_MODEL, 1.0f,
                                           out, (unsigned short*)0,
                                           (unsigned short*)0, x);

  // mlp pre-norm
  rmsnorm_bf16_k<<<M, 256, 0, stream>>>(out, mnw, hb);

  // router (one wave per token)
  router_k<<<(M + 7) / 8, 256, 0, stream>>>(hb, rw, fill, lists, slots, gatesb,
                                            cntf, psum, M);

  // experts
  dim3 g1((M / 16) * (FF_DIM / BN), N_EXP);
  moe1_k<<<g1, 256, 0, stream>>>(hb, w1t, b1, lists, slots, fill, zrow, act, M);
  dim3 g2((M / 16) * (D_MODEL / BN), N_EXP);
  moe2_k<<<g2, 256, 0, stream>>>(act, w2t, b2, lists, slots, gatesb, fill, zrow, out, M);

  // balancing loss + counts tail
  loss_k<<<1, 32, 0, stream>>>(cntf, psum, out + (size_t)M * D_MODEL, M);
}
